// ComnetModel_52097953300871
// MI455X (gfx1250) — matrix operand order repaired and checked by reading, not compiled, not run on hardware
//
#include <hip/hip_runtime.h>
#include <hip/hip_bf16.h>

// ---------------------------------------------------------------------------
// RouteNet-style GNN on MI455X (gfx1250), wave32 + V_WMMA_F32_16X16X32_F16.
// State kept in f32; matmul operands fed as f16 with f32 WMMA accumulation.
// One wave == one 16-row tile; EXEC is all-ones around every WMMA.
// ---------------------------------------------------------------------------

#define N_LINKS  20000
#define N_PATHS  100000
#define PATH_LEN 8
#define LINK_DIM 32
#define PATH_DIM 32
#define READOUT  256
#define T_ITERS  3

typedef __attribute__((ext_vector_type(16))) _Float16 v16h;
typedef __attribute__((ext_vector_type(8)))  _Float16 v8h;
typedef __attribute__((ext_vector_type(8)))  float    v8f;

__device__ __forceinline__ v8f splat8(float x) {
  v8f v;
#pragma unroll
  for (int i = 0; i < 8; ++i) v[i] = x;
  return v;
}

// Build a 16-half A/B fragment from two 8-half (16B) aligned chunks.
__device__ __forceinline__ v16h frag16(const _Float16* p0, const _Float16* p1) {
  v8h a = *(const v8h*)p0;
  v8h b = *(const v8h*)p1;
  v16h r;
#pragma unroll
  for (int i = 0; i < 8; ++i) { r[i] = a[i]; r[i + 8] = b[i]; }
  return r;
}

// Same, but converting from f32 source on the fly.
__device__ __forceinline__ v16h frag16_f32(const float* p0, const float* p1) {
  v16h r;
#pragma unroll
  for (int i = 0; i < 8; ++i) {
    r[i]     = (_Float16)p0[i];
    r[i + 8] = (_Float16)p1[i];
  }
  return r;
}

__device__ __forceinline__ v8f wmma_f16(v16h a, v16h b, v8f c) {
  // D = A(16x32 f16) * B(32x16 f16) + C(16x16 f32)
  return __builtin_amdgcn_wmma_f32_16x16x32_f16(
      /*neg_a=*/false, a, /*neg_b=*/false, b,
      /*c_mod=*/(short)0, c, /*reuse_a=*/false, /*reuse_b=*/false);
}

__device__ __forceinline__ float sigmoidf_(float x) {
  return 1.0f / (1.0f + __expf(-x));
}

// tanh via TRANS-pipe exp/rcp: tanh(x) = 1 - 2/(exp(2x)+1)
__device__ __forceinline__ float tanhf_(float x) {
  return 1.0f - 2.0f / (__expf(2.0f * x) + 1.0f);
}

__device__ __forceinline__ float seluf_(float x) {
  const float a = 1.6732632423543772f;
  const float s = 1.0507009873554805f;
  return x > 0.0f ? s * x : s * a * (__expf(x) - 1.0f);
}

// --------------------------- small setup kernels ---------------------------

__global__ void k_cvt_f16(const float* __restrict__ s, _Float16* __restrict__ d, int n) {
  int i = blockIdx.x * blockDim.x + threadIdx.x;
  if (i < n) d[i] = (_Float16)s[i];
}

__global__ void k_init_path(const float* __restrict__ traffic, float* __restrict__ ps) {
  int i = blockIdx.x * blockDim.x + threadIdx.x;
  if (i < N_PATHS * 32) ps[i] = ((i & 31) == 0) ? traffic[i >> 5] : 0.0f;
}

__global__ void k_init_link(const float* __restrict__ cap,
                            float* __restrict__ lf32, _Float16* __restrict__ lf16) {
  int i = blockIdx.x * blockDim.x + threadIdx.x;
  if (i < N_LINKS * 32) {
    float v = ((i & 31) == 0) ? cap[i >> 5] : 0.0f;
    lf32[i] = v;
    lf16[i] = (_Float16)v;
  }
}

// --------------------------- path GRU + scatter ----------------------------
// One wave handles 16 paths through all 8 GRU steps.
// Per step: 12 WMMAs (6 N-tiles for x@Wx, 6 for h@Wh), gates on f32
// accumulators (D layout), scatter each step's output into agg with
// global_atomic_add_f32, then LDS transpose of h for the next h@Wh.

__global__ void __launch_bounds__(32) k_path(
    float* __restrict__ path_state,        // [N_PATHS][32] f32 in/out
    const _Float16* __restrict__ link16,   // [N_LINKS][32] f16
    const int* __restrict__ links,         // [E]
    const _Float16* __restrict__ Wx,       // [32][96] f16
    const _Float16* __restrict__ Wh,       // [32][96] f16
    const float* __restrict__ b,           // [96]
    float* __restrict__ agg)               // [N_LINKS][32] f32 (pre-zeroed)
{
  __shared__ __align__(16) _Float16 sh[16 * 32];

  const int lane = threadIdx.x;
  const int row  = lane & 15;
  const int half = lane >> 4;            // 0: K 0-7/16-23, 1: K 8-15/24-31
  const int base = blockIdx.x * 16;

  // Cache B fragments for Wx/Wh (6 column tiles each) + per-lane bias.
  v16h bx[6], bh[6];
  float bias[6];
#pragma unroll
  for (int n = 0; n < 6; ++n) {
    const _Float16* px = Wx + 16 * n + row;
    const _Float16* ph = Wh + 16 * n + row;
#pragma unroll
    for (int i = 0; i < 8; ++i) {
      bx[n][i]     = px[(8 * half + i) * 96];
      bx[n][i + 8] = px[(16 + 8 * half + i) * 96];
      bh[n][i]     = ph[(8 * half + i) * 96];
      bh[n][i + 8] = ph[(16 + 8 * half + i) * 96];
    }
    bias[n] = b[16 * n + row];
  }

  // Load h in C/D layout: element (M = j + 8*half, N = 16*k + row).
  float hD[2][8];
#pragma unroll
  for (int k = 0; k < 2; ++k)
#pragma unroll
    for (int j = 0; j < 8; ++j)
      hD[k][j] = path_state[(size_t)(base + j + 8 * half) * 32 + 16 * k + row];

  // Transpose h into A layout via LDS.
#pragma unroll
  for (int k = 0; k < 2; ++k)
#pragma unroll
    for (int j = 0; j < 8; ++j)
      sh[(j + 8 * half) * 32 + 16 * k + row] = (_Float16)hD[k][j];
  __syncthreads();
  v16h hfrag = frag16(sh + row * 32 + 8 * half, sh + row * 32 + 8 * half + 16);

  for (int t = 0; t < PATH_LEN; ++t) {
    // x_t = link_state[links[path*8 + t]]  -> A fragment
    const int mylink = links[(size_t)(base + row) * PATH_LEN + t];
    const _Float16* lp = link16 + (size_t)mylink * 32 + 8 * half;
    v16h xfrag = frag16(lp, lp + 16);

    v8f gx[6], gh[6];
#pragma unroll
    for (int n = 0; n < 6; ++n) {
      gx[n] = wmma_f16(xfrag, bx[n], splat8(bias[n]));  // x@Wx + b
      gh[n] = wmma_f16(hfrag, bh[n], splat8(0.0f));     // h@Wh
    }

    // Gates: z = tiles 0-1, r = tiles 2-3, c = tiles 4-5.
    float hnew[2][8];
#pragma unroll
    for (int k = 0; k < 2; ++k)
#pragma unroll
      for (int j = 0; j < 8; ++j) {
        float z = sigmoidf_(gx[k][j]     + gh[k][j]);
        float r = sigmoidf_(gx[2 + k][j] + gh[2 + k][j]);
        float c = tanhf_(gx[4 + k][j] + r * gh[4 + k][j]);
        hnew[k][j] = z * hD[k][j] + (1.0f - z) * c;
      }

    // Unsorted segment-sum: atomically scatter this step's output rows.
#pragma unroll
    for (int j = 0; j < 8; ++j) {
      const int lk = links[(size_t)(base + j + 8 * half) * PATH_LEN + t];
      atomicAdd(agg + (size_t)lk * 32 + row,      hnew[0][j]);
      atomicAdd(agg + (size_t)lk * 32 + 16 + row, hnew[1][j]);
    }

    // h <- h_new ; rebuild A fragment through LDS transpose.
    __syncthreads();
#pragma unroll
    for (int k = 0; k < 2; ++k)
#pragma unroll
      for (int j = 0; j < 8; ++j) {
        hD[k][j] = hnew[k][j];
        sh[(j + 8 * half) * 32 + 16 * k + row] = (_Float16)hnew[k][j];
      }
    __syncthreads();
    hfrag = frag16(sh + row * 32 + 8 * half, sh + row * 32 + 8 * half + 16);
  }

#pragma unroll
  for (int k = 0; k < 2; ++k)
#pragma unroll
    for (int j = 0; j < 8; ++j)
      path_state[(size_t)(base + j + 8 * half) * 32 + 16 * k + row] = hD[k][j];
}

// ------------------------------- link GRU ----------------------------------

__global__ void __launch_bounds__(32) k_edge(
    const float* __restrict__ agg,       // [N_LINKS][32]
    float* __restrict__ lf32,            // [N_LINKS][32] in/out
    _Float16* __restrict__ lf16,         // [N_LINKS][32] out (for next gather)
    const _Float16* __restrict__ Wx,     // [32][96] f16
    const _Float16* __restrict__ Wh,     // [32][96] f16
    const float* __restrict__ b)         // [96]
{
  const int lane = threadIdx.x;
  const int row  = lane & 15;
  const int half = lane >> 4;
  const int base = blockIdx.x * 16;

  v16h bx[6], bh[6];
  float bias[6];
#pragma unroll
  for (int n = 0; n < 6; ++n) {
    const _Float16* px = Wx + 16 * n + row;
    const _Float16* ph = Wh + 16 * n + row;
#pragma unroll
    for (int i = 0; i < 8; ++i) {
      bx[n][i]     = px[(8 * half + i) * 96];
      bx[n][i + 8] = px[(16 + 8 * half + i) * 96];
      bh[n][i]     = ph[(8 * half + i) * 96];
      bh[n][i + 8] = ph[(16 + 8 * half + i) * 96];
    }
    bias[n] = b[16 * n + row];
  }

  const float* ap = agg  + (size_t)(base + row) * 32 + 8 * half;
  const float* hp = lf32 + (size_t)(base + row) * 32 + 8 * half;
  v16h afrag = frag16_f32(ap, ap + 16);
  v16h hfrag = frag16_f32(hp, hp + 16);

  float hD[2][8];
#pragma unroll
  for (int k = 0; k < 2; ++k)
#pragma unroll
    for (int j = 0; j < 8; ++j)
      hD[k][j] = lf32[(size_t)(base + j + 8 * half) * 32 + 16 * k + row];

  v8f gx[6], gh[6];
#pragma unroll
  for (int n = 0; n < 6; ++n) {
    gx[n] = wmma_f16(afrag, bx[n], splat8(bias[n]));
    gh[n] = wmma_f16(hfrag, bh[n], splat8(0.0f));
  }

#pragma unroll
  for (int k = 0; k < 2; ++k)
#pragma unroll
    for (int j = 0; j < 8; ++j) {
      float z = sigmoidf_(gx[k][j]     + gh[k][j]);
      float r = sigmoidf_(gx[2 + k][j] + gh[2 + k][j]);
      float c = tanhf_(gx[4 + k][j] + r * gh[4 + k][j]);
      float hn = z * hD[k][j] + (1.0f - z) * c;
      size_t o = (size_t)(base + j + 8 * half) * 32 + 16 * k + row;
      lf32[o] = hn;
      lf16[o] = (_Float16)hn;
    }
}

// ------------------------------- readout -----------------------------------
// h1 = selu(ps @ W1 + b1)  (32 -> 256), staged as f16 in LDS
// h2 = selu(h1 @ W2 + b2)  (256 -> 256), fused column-wise with
// out = h2 @ W3 + b3       (256 -> 1), reduced across lanes via ds_add_f32.

__global__ void __launch_bounds__(32) k_readout(
    const float* __restrict__ ps,       // [N_PATHS][32]
    const _Float16* __restrict__ W1,    // [32][256] f16
    const float* __restrict__ b1,       // [256]
    const _Float16* __restrict__ W2,    // [256][256] f16
    const float* __restrict__ b2,       // [256]
    const float* __restrict__ W3,       // [256]
    const float* __restrict__ b3,       // [1]
    float* __restrict__ out)            // [N_PATHS]
{
  __shared__ __align__(16) _Float16 sh1[16 * 256];   // h1 tile, f16
  __shared__ float sred[16];

  const int lane = threadIdx.x;
  const int row  = lane & 15;
  const int half = lane >> 4;
  const int base = blockIdx.x * 16;

  const float* pp = ps + (size_t)(base + row) * 32 + 8 * half;
  v16h pfrag = frag16_f32(pp, pp + 16);

  // ---- layer 1: 16 N-tiles of 32->256 ----
  for (int n = 0; n < 16; ++n) {
    const _Float16* w1p = W1 + 16 * n + row;
    v16h w1f;
#pragma unroll
    for (int i = 0; i < 8; ++i) {
      w1f[i]     = w1p[(8 * half + i) * 256];
      w1f[i + 8] = w1p[(16 + 8 * half + i) * 256];
    }
    v8f acc = wmma_f16(pfrag, w1f, splat8(b1[16 * n + row]));
#pragma unroll
    for (int j = 0; j < 8; ++j)
      sh1[(j + 8 * half) * 256 + 16 * n + row] = (_Float16)seluf_(acc[j]);
  }
  if (lane < 16) sred[lane] = 0.0f;
  __syncthreads();

  // ---- layer 2 fused with W3 dot ----
  float partial[8];
#pragma unroll
  for (int j = 0; j < 8; ++j) partial[j] = 0.0f;

  for (int n = 0; n < 16; ++n) {
    v8f acc = splat8(b2[16 * n + row]);
    for (int c = 0; c < 8; ++c) {
      const _Float16* hp = sh1 + row * 256 + 32 * c + 8 * half;
      v16h hf = frag16(hp, hp + 16);
      const _Float16* w2p = W2 + (size_t)(32 * c) * 256 + 16 * n + row;
      v16h wf;
#pragma unroll
      for (int i = 0; i < 8; ++i) {
        wf[i]     = w2p[(8 * half + i) * 256];
        wf[i + 8] = w2p[(16 + 8 * half + i) * 256];
      }
      acc = wmma_f16(hf, wf, acc);
    }
    const float w3v = W3[16 * n + row];
#pragma unroll
    for (int j = 0; j < 8; ++j) partial[j] += seluf_(acc[j]) * w3v;
  }

#pragma unroll
  for (int j = 0; j < 8; ++j) atomicAdd(&sred[j + 8 * half], partial[j]);
  __syncthreads();
  if (lane < 16) out[base + lane] = sred[lane] + b3[0];
}

// ------------------------------- launcher ----------------------------------

extern "C" void kernel_launch(void* const* d_in, const int* in_sizes, int n_in,
                              void* d_out, int out_size, void* d_ws, size_t ws_size,
                              hipStream_t stream) {
  (void)in_sizes; (void)n_in; (void)out_size; (void)ws_size;

  const float* cap     = (const float*)d_in[0];
  const float* traffic = (const float*)d_in[1];
  const int*   links   = (const int*)d_in[2];
  // d_in[3] paths, d_in[4] seqs: implied by the fixed (path, hop) layout.
  const float* Wx_path = (const float*)d_in[5];
  const float* Wh_path = (const float*)d_in[6];
  const float* b_path  = (const float*)d_in[7];
  const float* Wx_edge = (const float*)d_in[8];
  const float* Wh_edge = (const float*)d_in[9];
  const float* b_edge  = (const float*)d_in[10];
  const float* W1      = (const float*)d_in[11];
  const float* b1      = (const float*)d_in[12];
  const float* W2      = (const float*)d_in[13];
  const float* b2      = (const float*)d_in[14];
  const float* W3      = (const float*)d_in[15];
  const float* b3      = (const float*)d_in[16];
  float* out = (float*)d_out;

  // ---- workspace carve-up (all offsets 256B aligned) ----
  char* w = (char*)d_ws;
  size_t off = 0;
  auto take = [&](size_t bytes) -> char* {
    char* p = w + off;
    off = (off + bytes + 255) & ~(size_t)255;
    return p;
  };
  float*     lf32 = (float*)take((size_t)N_LINKS * 32 * 4);
  _Float16*  lf16 = (_Float16*)take((size_t)N_LINKS * 32 * 2);
  float*     ps   = (float*)take((size_t)N_PATHS * 32 * 4);
  float*     agg  = (float*)take((size_t)N_LINKS * 32 * 4);
  _Float16*  Wxp  = (_Float16*)take(32 * 96 * 2);
  _Float16*  Whp  = (_Float16*)take(32 * 96 * 2);
  _Float16*  Wxe  = (_Float16*)take(32 * 96 * 2);
  _Float16*  Whe  = (_Float16*)take(32 * 96 * 2);
  _Float16*  W1h  = (_Float16*)take(32 * 256 * 2);
  _Float16*  W2h  = (_Float16*)take(256 * 256 * 2);

  // ---- weight conversion + state init ----
  k_cvt_f16<<<(3072 + 255) / 256, 256, 0, stream>>>(Wx_path, Wxp, 3072);
  k_cvt_f16<<<(3072 + 255) / 256, 256, 0, stream>>>(Wh_path, Whp, 3072);
  k_cvt_f16<<<(3072 + 255) / 256, 256, 0, stream>>>(Wx_edge, Wxe, 3072);
  k_cvt_f16<<<(3072 + 255) / 256, 256, 0, stream>>>(Wh_edge, Whe, 3072);
  k_cvt_f16<<<(8192 + 255) / 256, 256, 0, stream>>>(W1, W1h, 8192);
  k_cvt_f16<<<(65536 + 255) / 256, 256, 0, stream>>>(W2, W2h, 65536);
  k_init_path<<<(N_PATHS * 32 + 255) / 256, 256, 0, stream>>>(traffic, ps);
  k_init_link<<<(N_LINKS * 32 + 255) / 256, 256, 0, stream>>>(cap, lf32, lf16);

  // ---- T message-passing rounds ----
  for (int it = 0; it < T_ITERS; ++it) {
    (void)hipMemsetAsync(agg, 0, (size_t)N_LINKS * 32 * 4, stream);
    k_path<<<N_PATHS / 16, 32, 0, stream>>>(ps, lf16, links, Wxp, Whp, b_path, agg);
    k_edge<<<N_LINKS / 16, 32, 0, stream>>>(agg, lf32, lf16, Wxe, Whe, b_edge);
  }

  // ---- readout MLP ----
  k_readout<<<N_PATHS / 16, 32, 0, stream>>>(ps, W1h, b1, W2h, b2, W3, b3, out);
}
